// LSTM_autoencoder_with_classification_27315992003081
// MI455X (gfx1250) — matrix operand order repaired and checked by reading, not compiled, not run on hardware
//
#include <hip/hip_runtime.h>
#include <hip/hip_bf16.h>

// ---------------------------------------------------------------------------
// LSTM autoencoder + classifier for MI455X (gfx1250, wave32, WMMA).
//
// Design: fp16 weights/activations with fp32 accumulation via
// v_wmma_f32_16x16x32_f16. One fused kernel per LSTM layer-step computes
// i/f/g/o gate tiles + activations + cell update. Weights converted to f16
// once per call (16 MB -> resident in 192 MB L2 across all 512 steps).
// Epilogue uses hardware transcendentals (v_exp_f32 / v_rcp_f32) to keep the
// per-step dependent chain short (2048 dependent kernels in the sequence).
// ---------------------------------------------------------------------------

typedef _Float16 v8h  __attribute__((ext_vector_type(8)));
typedef _Float16 v16h __attribute__((ext_vector_type(16)));
typedef float    v8f  __attribute__((ext_vector_type(8)));

#define HID   512
#define BATCH 128
#define SEQ   512
#define INP   256
#define NCLS  10
#define BH    (BATCH * HID)

// ---- WMMA fragment loaders (per CDNA5 ISA 16-bit A/B layouts) --------------
// A (16x32, MxK): lane L<16 -> row L, K {0..7,16..23}; lane L>=16 -> row L-16,
// K {8..15,24..31} (offsets within the K=32 chunk).
__device__ __forceinline__ v16h load_a_frag(const _Float16* __restrict__ row,
                                            int kc, int hi) {
  const _Float16* p = row + kc + (hi ? 8 : 0);
  v8h a0 = *(const v8h*)(p);
  v8h a1 = *(const v8h*)(p + 16);
  v16h r;
#pragma unroll
  for (int i = 0; i < 8; ++i) { r[i] = a0[i]; r[i + 8] = a1[i]; }
  return r;
}

// B (32x16, KxN) with B = W^T: lane L<16 -> W row (N=L), K {0..7,16..23};
// lane L>=16 -> W row (N=L-16), K {8..15,24..31} (same K order as A).
__device__ __forceinline__ v16h load_b_frag(const _Float16* __restrict__ row,
                                            int kc, int hi) {
  const _Float16* p = row + kc + (hi ? 8 : 0);
  v8h b0 = *(const v8h*)(p);
  v8h b1 = *(const v8h*)(p + 16);
  v16h r;
#pragma unroll
  for (int i = 0; i < 8; ++i) { r[i] = b0[i]; r[i + 8] = b1[i]; }
  return r;
}

// Fast activations on the hardware transcendental pipe:
//   sigmoid(x) = rcp(1 + exp(-x))          (v_exp_f32 + v_rcp_f32)
//   tanh(x)    = 1 - 2*rcp(exp(2x) + 1)    (v_exp_f32 + v_rcp_f32)
__device__ __forceinline__ float fast_sigmoid(float x) {
  return __builtin_amdgcn_rcpf(1.0f + __expf(-x));
}
__device__ __forceinline__ float fast_tanh(float x) {
  return 1.0f - 2.0f * __builtin_amdgcn_rcpf(__expf(2.0f * x) + 1.0f);
}

// ---- utility kernels -------------------------------------------------------
__global__ void cvt_f32_to_f16_kernel(const float* __restrict__ in,
                                      _Float16* __restrict__ out, int n) {
  int i = blockIdx.x * blockDim.x + threadIdx.x;
  int stride = gridDim.x * blockDim.x;
  for (; i < n; i += stride) out[i] = (_Float16)in[i];
}

__global__ void fill_zero_kernel(unsigned int* __restrict__ p, int ndw) {
  int i = blockIdx.x * blockDim.x + threadIdx.x;
  int stride = gridDim.x * blockDim.x;
  for (; i < ndw; i += stride) p[i] = 0u;
}

// ---- fused LSTM layer-step -------------------------------------------------
// gates = xA @ Wih^T + hA @ Whh^T + b ; gate order [i,f,g,o] each HID wide.
// One wave per (m_tile, n_tile): 4 gate accumulators share each A fragment.
// grid = (BATCH/16) * (HID/16) = 8 * 32 = 256 waves, blockDim = 32.
__global__ void lstm_step_kernel(const _Float16* __restrict__ xA, int lda, int Kx,
                                 const _Float16* __restrict__ hA,
                                 const _Float16* __restrict__ Wih,
                                 const _Float16* __restrict__ Whh,
                                 const float* __restrict__ bias,
                                 const float* __restrict__ c_in,
                                 _Float16* __restrict__ h_out,
                                 float* __restrict__ c_out) {
  const int lane = threadIdx.x & 31;
  const int wid  = blockIdx.x;
  const int mt = wid >> 5;          // 0..7
  const int nt = wid & 31;          // 0..31
  const int hi = lane >> 4;
  const int ln = lane & 15;
  const int nG = nt * 16 + ln;      // column within one gate (0..511)

  v8f ai = {}, af = {}, ag = {}, ao = {};

  // phase 1: input contribution  x @ Wih^T
  {
    const _Float16* arow = xA + (size_t)(mt * 16 + ln) * lda;
    const _Float16* wi = Wih + (size_t)(0 * HID + nG) * Kx;
    const _Float16* wf = Wih + (size_t)(1 * HID + nG) * Kx;
    const _Float16* wg = Wih + (size_t)(2 * HID + nG) * Kx;
    const _Float16* wo = Wih + (size_t)(3 * HID + nG) * Kx;
    for (int kc = 0; kc < Kx; kc += 32) {
      v16h a  = load_a_frag(arow, kc, hi);
      v16h bi = load_b_frag(wi, kc, hi);
      v16h bf = load_b_frag(wf, kc, hi);
      v16h bg = load_b_frag(wg, kc, hi);
      v16h bo = load_b_frag(wo, kc, hi);
      ai = __builtin_amdgcn_wmma_f32_16x16x32_f16(false, a, false, bi, (short)0, ai, false, false);
      af = __builtin_amdgcn_wmma_f32_16x16x32_f16(false, a, false, bf, (short)0, af, false, false);
      ag = __builtin_amdgcn_wmma_f32_16x16x32_f16(false, a, false, bg, (short)0, ag, false, false);
      ao = __builtin_amdgcn_wmma_f32_16x16x32_f16(false, a, false, bo, (short)0, ao, false, false);
    }
  }
  // phase 2: recurrent contribution  h @ Whh^T
  {
    const _Float16* arow = hA + (size_t)(mt * 16 + ln) * HID;
    const _Float16* wi = Whh + (size_t)(0 * HID + nG) * HID;
    const _Float16* wf = Whh + (size_t)(1 * HID + nG) * HID;
    const _Float16* wg = Whh + (size_t)(2 * HID + nG) * HID;
    const _Float16* wo = Whh + (size_t)(3 * HID + nG) * HID;
    for (int kc = 0; kc < HID; kc += 32) {
      v16h a  = load_a_frag(arow, kc, hi);
      v16h bi = load_b_frag(wi, kc, hi);
      v16h bf = load_b_frag(wf, kc, hi);
      v16h bg = load_b_frag(wg, kc, hi);
      v16h bo = load_b_frag(wo, kc, hi);
      ai = __builtin_amdgcn_wmma_f32_16x16x32_f16(false, a, false, bi, (short)0, ai, false, false);
      af = __builtin_amdgcn_wmma_f32_16x16x32_f16(false, a, false, bf, (short)0, af, false, false);
      ag = __builtin_amdgcn_wmma_f32_16x16x32_f16(false, a, false, bg, (short)0, ag, false, false);
      ao = __builtin_amdgcn_wmma_f32_16x16x32_f16(false, a, false, bo, (short)0, ao, false, false);
    }
  }

  // epilogue: bias + activations + cell update (C layout: vgpr r -> row
  // mt*16 + r + hi*8, col nG)
  const float b_i = bias[0 * HID + nG];
  const float b_f = bias[1 * HID + nG];
  const float b_g = bias[2 * HID + nG];
  const float b_o = bias[3 * HID + nG];
#pragma unroll
  for (int r = 0; r < 8; ++r) {
    const int m = mt * 16 + r + hi * 8;
    const float iv = fast_sigmoid(ai[r] + b_i);
    const float fv = fast_sigmoid(af[r] + b_f);
    const float gv = fast_tanh(ag[r] + b_g);
    const float ov = fast_sigmoid(ao[r] + b_o);
    const float c_old = c_in[(size_t)m * HID + nG];
    const float cn = fv * c_old + iv * gv;
    const float hn = ov * fast_tanh(cn);
    c_out[(size_t)m * HID + nG] = cn;
    h_out[(size_t)m * HID + nG] = (_Float16)hn;
  }
}

// ---- generic out = A @ W^T + bias (M fixed = 128) --------------------------
// grid = 8 * n_tiles waves, blockDim = 32. n_valid guards tails (NCLS=10).
__global__ void gemm_wtb_kernel(const _Float16* __restrict__ A, int lda,
                                const _Float16* __restrict__ W, int K,
                                const float* __restrict__ bias,
                                float* __restrict__ outF32,
                                _Float16* __restrict__ outF16,
                                int ldc, int n_valid, int relu, int n_tiles) {
  const int lane = threadIdx.x & 31;
  const int wid  = blockIdx.x;
  const int mt = wid / n_tiles;
  const int nt = wid % n_tiles;
  const int hi = lane >> 4;
  const int ln = lane & 15;
  const int nG = nt * 16 + ln;
  const bool rowok = nG < n_valid;

  const _Float16* arow = A + (size_t)(mt * 16 + ln) * lda;
  const _Float16* wrow = W + (size_t)(rowok ? nG : 0) * K;

  v8f acc = {};
  for (int kc = 0; kc < K; kc += 32) {
    v16h a = load_a_frag(arow, kc, hi);
    v16h b = {};
    if (rowok) b = load_b_frag(wrow, kc, hi);
    acc = __builtin_amdgcn_wmma_f32_16x16x32_f16(false, a, false, b, (short)0, acc, false, false);
  }
  const float bv = rowok ? bias[nG] : 0.0f;
#pragma unroll
  for (int r = 0; r < 8; ++r) {
    const int m = mt * 16 + r + hi * 8;
    float v = acc[r] + bv;
    if (relu) v = fmaxf(v, 0.0f);
    if (rowok) {
      if (outF32) outF32[(size_t)m * ldc + nG] = v;
      else        outF16[(size_t)m * ldc + nG] = (_Float16)v;
    }
  }
}

// ---------------------------------------------------------------------------
extern "C" void kernel_launch(void* const* d_in, const int* in_sizes, int n_in,
                              void* d_out, int out_size, void* d_ws, size_t ws_size,
                              hipStream_t stream) {
  (void)in_sizes; (void)n_in; (void)out_size; (void)ws_size;

  // inputs (setup_inputs order)
  const float* x        = (const float*)d_in[0];
  const float* encWih0  = (const float*)d_in[1];
  const float* encWhh0  = (const float*)d_in[2];
  const float* enc_b0   = (const float*)d_in[3];
  const float* encWih1  = (const float*)d_in[4];
  const float* encWhh1  = (const float*)d_in[5];
  const float* enc_b1   = (const float*)d_in[6];
  const float* decWih0  = (const float*)d_in[7];
  const float* decWhh0  = (const float*)d_in[8];
  const float* dec_b0   = (const float*)d_in[9];
  const float* decWih1  = (const float*)d_in[10];
  const float* decWhh1  = (const float*)d_in[11];
  const float* dec_b1   = (const float*)d_in[12];
  const float* fc_W     = (const float*)d_in[13];
  const float* fc_b     = (const float*)d_in[14];
  const float* cls1_W   = (const float*)d_in[15];
  const float* cls1_b   = (const float*)d_in[16];
  const float* cls2_W   = (const float*)d_in[17];
  const float* cls2_b   = (const float*)d_in[18];

  float* dout = (float*)d_out;

  // ---- workspace carving (256B aligned) ----
  char* ws = (char*)d_ws;
  size_t off = 0;
  auto carve = [&](size_t bytes) -> void* {
    off = (off + 255) & ~(size_t)255;
    void* p = ws + off;
    off += bytes;
    return p;
  };

  _Float16* x16       = (_Float16*)carve((size_t)BATCH * SEQ * INP * 2);
  _Float16* encWih0_h = (_Float16*)carve((size_t)4 * HID * INP * 2);
  _Float16* encWhh0_h = (_Float16*)carve((size_t)4 * HID * HID * 2);
  _Float16* encWih1_h = (_Float16*)carve((size_t)4 * HID * HID * 2);
  _Float16* encWhh1_h = (_Float16*)carve((size_t)4 * HID * HID * 2);
  _Float16* decWih0_h = (_Float16*)carve((size_t)4 * HID * HID * 2);
  _Float16* decWhh0_h = (_Float16*)carve((size_t)4 * HID * HID * 2);
  _Float16* decWih1_h = (_Float16*)carve((size_t)4 * HID * HID * 2);
  _Float16* decWhh1_h = (_Float16*)carve((size_t)4 * HID * HID * 2);
  _Float16* fcW_h     = (_Float16*)carve((size_t)INP * HID * 2);
  _Float16* cls1W_h   = (_Float16*)carve((size_t)HID * HID * 2);
  _Float16* cls2W_h   = (_Float16*)carve((size_t)NCLS * HID * 2);
  _Float16* hpool     = (_Float16*)carve((size_t)8 * BH * 2);   // 8 x [B,H] f16
  float*    cpool     = (float*)   carve((size_t)8 * BH * 4);   // 8 x [B,H] f32
  _Float16* clsTmp    = (_Float16*)carve((size_t)BH * 2);

  _Float16* hE0[2] = {hpool + 0 * BH, hpool + 1 * BH};
  _Float16* hE1[2] = {hpool + 2 * BH, hpool + 3 * BH};
  _Float16* hD0[2] = {hpool + 4 * BH, hpool + 5 * BH};
  _Float16* hD1[2] = {hpool + 6 * BH, hpool + 7 * BH};
  float*    cE0[2] = {cpool + 0 * BH, cpool + 1 * BH};
  float*    cE1[2] = {cpool + 2 * BH, cpool + 3 * BH};
  float*    cD0[2] = {cpool + 4 * BH, cpool + 5 * BH};
  float*    cD1[2] = {cpool + 6 * BH, cpool + 7 * BH};

  // ---- one-time conversions (HBM read once; f16 weights live in L2) ----
  auto cvt = [&](const float* src, _Float16* dst, int n) {
    int grid = (n + 255) / 256;
    if (grid > 4096) grid = 4096;
    cvt_f32_to_f16_kernel<<<grid, 256, 0, stream>>>(src, dst, n);
  };
  cvt(x,       x16,       BATCH * SEQ * INP);
  cvt(encWih0, encWih0_h, 4 * HID * INP);
  cvt(encWhh0, encWhh0_h, 4 * HID * HID);
  cvt(encWih1, encWih1_h, 4 * HID * HID);
  cvt(encWhh1, encWhh1_h, 4 * HID * HID);
  cvt(decWih0, decWih0_h, 4 * HID * HID);
  cvt(decWhh0, decWhh0_h, 4 * HID * HID);
  cvt(decWih1, decWih1_h, 4 * HID * HID);
  cvt(decWhh1, decWhh1_h, 4 * HID * HID);
  cvt(fc_W,    fcW_h,     INP * HID);
  cvt(cls1_W,  cls1W_h,   HID * HID);
  cvt(cls2_W,  cls2W_h,   NCLS * HID);

  // zero initial states (h pool: 8*BH halfs = 4*BH dwords; c pool: 8*BH dwords)
  fill_zero_kernel<<<2048, 256, 0, stream>>>((unsigned int*)hpool, 4 * BH);
  fill_zero_kernel<<<2048, 256, 0, stream>>>((unsigned int*)cpool, 8 * BH);

  const int STEP_GRID = (BATCH / 16) * (HID / 16); // 256 waves

  // ---- encoder: 512 steps, 2 layers, double-buffered state ----
  int cur = 0;
  for (int t = 0; t < SEQ; ++t) {
    lstm_step_kernel<<<STEP_GRID, 32, 0, stream>>>(
        x16 + (size_t)t * INP, SEQ * INP, INP,
        hE0[cur], encWih0_h, encWhh0_h, enc_b0, cE0[cur],
        hE0[1 - cur], cE0[1 - cur]);
    lstm_step_kernel<<<STEP_GRID, 32, 0, stream>>>(
        hE0[1 - cur], HID, HID,
        hE1[cur], encWih1_h, encWhh1_h, enc_b1, cE1[cur],
        hE1[1 - cur], cE1[1 - cur]);
    cur ^= 1;
  }
  const _Float16* z = hE1[cur]; // final top-layer hidden state [B,H] f16

  // ---- decoder: constant input z; per-step fused FC into d_out ----
  int dc = 0;
  for (int t = 0; t < SEQ; ++t) {
    lstm_step_kernel<<<STEP_GRID, 32, 0, stream>>>(
        z, HID, HID,
        hD0[dc], decWih0_h, decWhh0_h, dec_b0, cD0[dc],
        hD0[1 - dc], cD0[1 - dc]);
    lstm_step_kernel<<<STEP_GRID, 32, 0, stream>>>(
        hD0[1 - dc], HID, HID,
        hD1[dc], decWih1_h, decWhh1_h, dec_b1, cD1[dc],
        hD1[1 - dc], cD1[1 - dc]);
    // decoded[:, t, :] = top @ fc_W^T + fc_b   (f32, strided into [B,S,IN])
    gemm_wtb_kernel<<<8 * (INP / 16), 32, 0, stream>>>(
        hD1[1 - dc], HID, fcW_h, HID, fc_b,
        dout + (size_t)t * INP, nullptr, SEQ * INP, INP, 0, INP / 16);
    dc ^= 1;
  }
  const _Float16* last = hD1[dc];

  // ---- classifier: relu(last @ cls1^T + b1) @ cls2^T + b2 ----
  gemm_wtb_kernel<<<8 * (HID / 16), 32, 0, stream>>>(
      last, HID, cls1W_h, HID, cls1_b,
      nullptr, clsTmp, HID, HID, 1, HID / 16);
  gemm_wtb_kernel<<<8 * 1, 32, 0, stream>>>(
      clsTmp, HID, cls2W_h, HID, cls2_b,
      dout + (size_t)BATCH * SEQ * INP, nullptr, NCLS, NCLS, 0, 1);
}
